// MemoryGraph_75874892251489
// MI455X (gfx1250) — matrix operand order repaired and checked by reading, not compiled, not run on hardware
//
#include <hip/hip_runtime.h>
#include <hip/hip_bf16.h>
#include <math.h>

typedef __attribute__((ext_vector_type(16))) __bf16 v16bf;
typedef __attribute__((ext_vector_type(8)))  float  v8f;

#define BSZ   16
#define NCC   64
#define NN    256
#define DD    64
#define HSS   256
#define HMM   256

__device__ __forceinline__ v8f wmma_bf16(v16bf a, v16bf b, v8f c) {
  // D = A(16x32 bf16) * B(32x16 bf16) + C(16x16 f32)
  return __builtin_amdgcn_wmma_f32_16x16x32_bf16(false, a, false, b, (short)0, c, false, false);
}

// ---------------------------------------------------------------------------
// Kernel 0: one-shot f32 -> bf16 pre-pack of the four shared weight matrices.
// They are L2-resident (192MB L2) and reused by all 1024 blocks of the fused
// kernel, so packing once removes all per-fragment cvt VALU from inner loops.
// ---------------------------------------------------------------------------
__global__ __launch_bounds__(256) void pack_weights(
    const float* __restrict__ sw1, const float* __restrict__ sw2,
    const float* __restrict__ mw1, const float* __restrict__ mw2,
    __bf16* __restrict__ o1, __bf16* __restrict__ o2,
    __bf16* __restrict__ o3, __bf16* __restrict__ o4) {
  int i = blockIdx.x * 256 + threadIdx.x;
  int stride = gridDim.x * 256;
  for (int j = i; j < HSS * 128; j += stride) o1[j] = (__bf16)sw1[j];
  for (int j = i; j < DD * HSS;  j += stride) o2[j] = (__bf16)sw2[j];
  for (int j = i; j < HMM * DD;  j += stride) o3[j] = (__bf16)mw1[j];
  for (int j = i; j < DD * HMM;  j += stride) o4[j] = (__bf16)mw2[j];
}

// ---------------------------------------------------------------------------
// Kernel 1: per-cell modulation MLP -> decay (B,NC,N) and bgate (B,NC,4)
// ---------------------------------------------------------------------------
__global__ __launch_bounds__(64) void mod_kernel(
    const float* __restrict__ cc,   const float* __restrict__ dlog,
    const float* __restrict__ bglog,const float* __restrict__ mw1,
    const float* __restrict__ mb1,  const float* __restrict__ mw2,
    const float* __restrict__ mb2,  float* __restrict__ decay,
    float* __restrict__ bgate) {
  int bc = blockIdx.x;
  int c  = bc & (NCC - 1);
  int t  = threadIdx.x;            // 0..63
  __shared__ float hid[64];
  __shared__ float mod[8];
  const float* ccp = cc + (size_t)bc * DD;
  float s = mb1[c * 64 + t];
  for (int d = 0; d < DD; ++d)
    s += ccp[d] * mw1[((size_t)c * 64 + d) * 64 + t];
  hid[t] = tanhf(s);
  __syncthreads();
  if (t < 5) {
    float m = mb2[c * 5 + t];
    for (int k = 0; k < 64; ++k) m += hid[k] * mw2[((size_t)c * 64 + k) * 5 + t];
    mod[t] = m;
  }
  __syncthreads();
  float m0 = mod[0];
  for (int n = t; n < NN; n += 64)
    decay[(size_t)bc * NN + n] = 1.f / (1.f + expf(-(dlog[(size_t)bc * NN + n] + m0)));
  if (t < 4)
    bgate[bc * 4 + t] = 1.f / (1.f + expf(-(bglog[bc * 4 + t] + mod[1 + t])));
}

// ---------------------------------------------------------------------------
// Kernel 2: received[b,c] = W[b,c](256x256) @ msg[b,c](256x64), bf16 WMMA.
// msg staged transposed in LDS (bf16) so B-fragments are contiguous b128 reads.
// All four B-fragments are loaded before the WMMA chain so the ds_load_b128s
// issue as a clause and the compiler can use partial dscnt waits; W prefetch
// is unconditional (speculative prefetch past end is dropped by hardware).
// ---------------------------------------------------------------------------
__global__ __launch_bounds__(512) void recv_gemm(
    const float* __restrict__ W, const float* __restrict__ msg,
    float* __restrict__ recv) {
  int bc = blockIdx.x;
  const float* Wp = W   + (size_t)bc * NN * NN;
  const float* Mp = msg + (size_t)bc * NN * DD;
  float*       Rp = recv+ (size_t)bc * NN * DD;
  __shared__ __bf16 msgT[DD][NN + 8];     // msgT[d][k] = msg[k][d]
  for (int i = threadIdx.x; i < NN * DD; i += 512) {
    int n = i >> 6, d = i & 63;
    msgT[d][n] = (__bf16)Mp[i];
  }
  __syncthreads();
  int wave = threadIdx.x >> 5;            // 0..15 -> 16-row strip of output
  int lane = threadIdx.x & 31;
  int l16  = lane & 15, g = lane >> 4;
  v8f acc[4];
  #pragma unroll
  for (int s = 0; s < 4; ++s)
    #pragma unroll
    for (int r = 0; r < 8; ++r) acc[s][r] = 0.f;
  const float* arow = Wp + (size_t)(wave * 16 + l16) * NN;
  for (int k0 = 0; k0 < NN; k0 += 32) {
    __builtin_prefetch(arow + k0 + 32 + g * 8, 0, 1);   // global_prefetch_b8
    // A fragment: contiguous K order (elt j -> K=16g+j), matching B
    v16bf a;
    const float* ap = arow + k0 + g * 16;
    #pragma unroll
    for (int j = 0; j < 8; ++j) { a[j] = (__bf16)ap[j]; a[j + 8] = (__bf16)ap[8 + j]; }
    // Load all 4 B fragments first (simultaneously live -> clause of ds loads)
    v16bf bfr[4];
    #pragma unroll
    for (int s = 0; s < 4; ++s) {
      // B fragment: lane = column, elt j -> K = k0 + 16g + j (contiguous in msgT)
      const __bf16* bp = &msgT[s * 16 + l16][k0 + g * 16];
      #pragma unroll
      for (int j = 0; j < 16; ++j) bfr[s][j] = bp[j];
    }
    #pragma unroll
    for (int s = 0; s < 4; ++s)
      acc[s] = wmma_bf16(a, bfr[s], acc[s]);
  }
  #pragma unroll
  for (int s = 0; s < 4; ++s)
    #pragma unroll
    for (int r = 0; r < 8; ++r)   // C layout: VGPR r -> row r (+8 for upper half-wave)
      Rp[(size_t)(wave * 16 + g * 8 + r) * DD + s * 16 + l16] = acc[s][r];
}

// ---------------------------------------------------------------------------
// Kernel 3: received += injection (rows 0..3) and gated border (rows 8..11)
// ---------------------------------------------------------------------------
__global__ __launch_bounds__(256) void add_inj_border(
    const float* __restrict__ x,  const float* __restrict__ msg,
    const float* __restrict__ iw, const float* __restrict__ ib,
    const int*   __restrict__ c2g,const float* __restrict__ bgate,
    float* __restrict__ recv) {
  int bc = blockIdx.x;
  int b = bc >> 6, c = bc & 63;
  int t = threadIdx.x;                     // 0..255
  int grp = c2g[c];
  const float* xr  = x  + (size_t)b * (NCC * DD) + c * DD;
  const float* iwr = iw + ((size_t)grp * 256 + t) * DD;
  float s = ib[grp * 256 + t];
  for (int i = 0; i < DD; ++i) s += xr[i] * iwr[i];
  recv[((size_t)bc * NN + (t >> 6)) * DD + (t & 63)] += s;

  int p = t >> 6, d = t & 63;
  int gy = c >> 3, gx = c & 7;
  int nb = -1, q = p ^ 1;
  if (p == 0 && gy > 0) nb = c - 8;
  if (p == 1 && gy < 7) nb = c + 8;
  if (p == 2 && gx > 0) nb = c - 1;
  if (p == 3 && gx < 7) nb = c + 1;
  if (nb >= 0) {
    float inc  = msg[(((size_t)b * NCC + nb) * NN + 8 + q) * DD + d];
    float gate = bgate[bc * 4 + p];
    recv[((size_t)bc * NN + 8 + p) * DD + d] += gate * inc;
  }
}

// ---------------------------------------------------------------------------
// Kernel 4: fused state + message update for one (b,c), 4 row-tiles of 64.
//   hid_s = tanh([recv|h] @ sw1^T + sb1)            (64x128)@(128x256)
//   cand  = tanh(hid_s @ sw2^T + sb2); h_new = lerp (64x256)@(256x64)
//   hid_m = tanh(h_new @ mw1^T + mb1)               (64x64) @(64x256)
//   msg_new = tanh(hid_m @ mw2^T + mb2) + nid       (64x256)@(256x64)
// Intermediates kept as bf16 in LDS; pre-packed bf16 weight B-fragments are
// single aligned 32B loads (2x global_load_b128) from L2 — zero cvt VALU.
// ---------------------------------------------------------------------------
__global__ __launch_bounds__(256) void fused_update(
    const float* __restrict__ recv, const float* __restrict__ h,
    const float* __restrict__ decay,
    const __bf16* __restrict__ sw1b, const float* __restrict__ sb1,
    const __bf16* __restrict__ sw2b, const float* __restrict__ sb2,
    const __bf16* __restrict__ mw1b, const float* __restrict__ mb1,
    const __bf16* __restrict__ mw2b, const float* __restrict__ mb2,
    const float* __restrict__ nid,
    float* __restrict__ rout, float* __restrict__ hout, float* __restrict__ mout) {
  int bc = blockIdx.x;
  int b = bc >> 6, c = bc & 63;
  const float* Rp = recv + (size_t)bc * NN * DD;
  const float* Hp = h    + (size_t)bc * NN * DD;
  const float* Dp = decay+ (size_t)bc * NN;
  __shared__ __bf16 hidb[64][HSS + 8];    // 33.8 KB: hid_s / hid_m tile
  __shared__ __bf16 hnewb[64][DD + 8];    //  9.2 KB: h_new tile (bf16)
  int wave = threadIdx.x >> 5, lane = threadIdx.x & 31;
  int l16 = lane & 15, g = lane >> 4;
  int rb = wave & 3;                       // 16-row sub-block within 64-row tile

  for (int t = 0; t < 4; ++t) {
    int r0 = t * 64;
    // ---- stage A: hid_s (64x256), each wave: rows 16*rb, 8 col-tiles ----
    {
      int cg = wave >> 2;
      int arow = r0 + rb * 16 + l16;
      const float* rr = Rp + (size_t)arow * DD;
      const float* hr = Hp + (size_t)arow * DD;
      v8f acc[8];
      #pragma unroll
      for (int i = 0; i < 8; ++i)
        #pragma unroll
        for (int r = 0; r < 8; ++r) acc[i][r] = 0.f;
      #pragma unroll
      for (int kk = 0; kk < 4; ++kk) {
        int k0 = kk * 32;
        const float* src = (kk < 2) ? rr : hr;   // K: [recv(0..63) | h(64..127)]
        int kb = k0 & 63;
        v16bf a;
        #pragma unroll
        for (int j = 0; j < 8; ++j) {
          a[j]     = (__bf16)src[kb + g * 16 + j];
          a[j + 8] = (__bf16)src[kb + g * 16 + 8 + j];
        }
        #pragma unroll
        for (int ct = 0; ct < 8; ++ct) {
          v16bf bfr = *(const v16bf*)(sw1b + (size_t)((cg * 8 + ct) * 16 + l16) * 128 + k0 + g * 16);
          acc[ct] = wmma_bf16(a, bfr, acc[ct]);
        }
      }
      #pragma unroll
      for (int ct = 0; ct < 8; ++ct) {
        int col = (cg * 8 + ct) * 16 + l16;
        float bias = sb1[col];
        #pragma unroll
        for (int r = 0; r < 8; ++r)
          hidb[rb * 16 + g * 8 + r][col] = (__bf16)tanhf(acc[ct][r] + bias);
      }
    }
    __syncthreads();
    // ---- stage B: cand (64x64) + h_new ----
    {
      int cp = wave >> 2;
      v8f acc[2];
      #pragma unroll
      for (int i = 0; i < 2; ++i)
        #pragma unroll
        for (int r = 0; r < 8; ++r) acc[i][r] = 0.f;
      #pragma unroll
      for (int kk = 0; kk < 8; ++kk) {
        int k0 = kk * 32;
        const __bf16* ap = &hidb[rb * 16 + l16][k0 + g * 16];
        v16bf a;
        #pragma unroll
        for (int j = 0; j < 8; ++j) { a[j] = ap[j]; a[j + 8] = ap[8 + j]; }
        #pragma unroll
        for (int u = 0; u < 2; ++u) {
          v16bf bfr = *(const v16bf*)(sw2b + (size_t)((cp * 2 + u) * 16 + l16) * HSS + k0 + g * 16);
          acc[u] = wmma_bf16(a, bfr, acc[u]);
        }
      }
      #pragma unroll
      for (int u = 0; u < 2; ++u) {
        int col = (cp * 2 + u) * 16 + l16;
        float bias = sb2[col];
        #pragma unroll
        for (int r = 0; r < 8; ++r) {
          int lr = rb * 16 + g * 8 + r;
          int n  = r0 + lr;
          float cand = tanhf(acc[u][r] + bias);
          float dcy  = Dp[n];
          float hv   = Hp[(size_t)n * DD + col];
          float hn   = dcy * hv + (1.f - dcy) * cand;
          hout[((size_t)bc * NN + n) * DD + col] = hn;
          hnewb[lr][col] = (__bf16)hn;
        }
      }
    }
    __syncthreads();
    // ---- stage C: hid_m (64x256) from h_new ----
    {
      int cg = wave >> 2;
      v8f acc[8];
      #pragma unroll
      for (int i = 0; i < 8; ++i)
        #pragma unroll
        for (int r = 0; r < 8; ++r) acc[i][r] = 0.f;
      #pragma unroll
      for (int kk = 0; kk < 2; ++kk) {
        int k0 = kk * 32;
        const __bf16* ap = &hnewb[rb * 16 + l16][k0 + g * 16];
        v16bf a;
        #pragma unroll
        for (int j = 0; j < 8; ++j) { a[j] = ap[j]; a[j + 8] = ap[8 + j]; }
        #pragma unroll
        for (int ct = 0; ct < 8; ++ct) {
          v16bf bfr = *(const v16bf*)(mw1b + (size_t)((cg * 8 + ct) * 16 + l16) * DD + k0 + g * 16);
          acc[ct] = wmma_bf16(a, bfr, acc[ct]);
        }
      }
      #pragma unroll
      for (int ct = 0; ct < 8; ++ct) {
        int col = (cg * 8 + ct) * 16 + l16;
        float bias = mb1[col];
        #pragma unroll
        for (int r = 0; r < 8; ++r)
          hidb[rb * 16 + g * 8 + r][col] = (__bf16)tanhf(acc[ct][r] + bias);
      }
    }
    __syncthreads();
    // ---- stage D: msg_new (64x64) + readout mean over rows 4..7 ----
    {
      int cp = wave >> 2;
      v8f acc[2];
      #pragma unroll
      for (int i = 0; i < 2; ++i)
        #pragma unroll
        for (int r = 0; r < 8; ++r) acc[i][r] = 0.f;
      #pragma unroll
      for (int kk = 0; kk < 8; ++kk) {
        int k0 = kk * 32;
        const __bf16* ap = &hidb[rb * 16 + l16][k0 + g * 16];
        v16bf a;
        #pragma unroll
        for (int j = 0; j < 8; ++j) { a[j] = ap[j]; a[j + 8] = ap[8 + j]; }
        #pragma unroll
        for (int u = 0; u < 2; ++u) {
          v16bf bfr = *(const v16bf*)(mw2b + (size_t)((cp * 2 + u) * 16 + l16) * HMM + k0 + g * 16);
          acc[u] = wmma_bf16(a, bfr, acc[u]);
        }
      }
      #pragma unroll
      for (int u = 0; u < 2; ++u) {
        int col = (cp * 2 + u) * 16 + l16;
        float bias = mb2[col];
        float rsum = 0.f;
        #pragma unroll
        for (int r = 0; r < 8; ++r) {
          int lr = rb * 16 + g * 8 + r;
          int n  = r0 + lr;
          float m = tanhf(acc[u][r] + bias) + nid[((size_t)c * NN + n) * DD + col];
          mout[((size_t)bc * NN + n) * DD + col] = m;
          if (t == 0 && rb == 0 && g == 0 && r >= 4) rsum += m;  // rows 4..7
        }
        if (t == 0 && rb == 0 && g == 0)
          rout[(size_t)b * 4096 + c * 64 + col] = rsum * 0.25f;
      }
    }
    __syncthreads();
  }
}

// ---------------------------------------------------------------------------
extern "C" void kernel_launch(void* const* d_in, const int* in_sizes, int n_in,
                              void* d_out, int out_size, void* d_ws, size_t ws_size,
                              hipStream_t stream) {
  (void)in_sizes; (void)n_in; (void)out_size; (void)ws_size;
  const float* x     = (const float*)d_in[0];
  const float* h     = (const float*)d_in[1];
  const float* msg   = (const float*)d_in[2];
  const float* W     = (const float*)d_in[3];
  const float* dlog  = (const float*)d_in[4];
  const float* cc    = (const float*)d_in[5];
  const float* bglog = (const float*)d_in[6];
  const float* nid   = (const float*)d_in[7];
  const float* sw1   = (const float*)d_in[8];
  const float* sb1   = (const float*)d_in[9];
  const float* sw2   = (const float*)d_in[10];
  const float* sb2   = (const float*)d_in[11];
  const float* mw1   = (const float*)d_in[12];
  const float* mb1   = (const float*)d_in[13];
  const float* mw2   = (const float*)d_in[14];
  const float* mb2   = (const float*)d_in[15];
  const float* iw    = (const float*)d_in[16];
  const float* ib    = (const float*)d_in[17];
  const float* modw1 = (const float*)d_in[18];
  const float* modb1 = (const float*)d_in[19];
  const float* modw2 = (const float*)d_in[20];
  const float* modb2 = (const float*)d_in[21];
  const int*   c2g   = (const int*)d_in[22];

  float* ws    = (float*)d_ws;
  float* recv  = ws;                                   // 16*64*256*64 f32
  float* decay = ws + (size_t)BSZ * NCC * NN * DD;     // 16*64*256 f32
  float* bgate = decay + (size_t)BSZ * NCC * NN;       // 16*64*4 f32
  __bf16* sw1b = (__bf16*)(bgate + (size_t)BSZ * NCC * 4);  // 32B-aligned
  __bf16* sw2b = sw1b + (size_t)HSS * 128;
  __bf16* mw1b = sw2b + (size_t)DD * HSS;
  __bf16* mw2b = mw1b + (size_t)HMM * DD;

  float* rout = (float*)d_out;                          // (16, 4096)
  float* hout = rout + (size_t)BSZ * NCC * DD;          // (16,64,256,64)
  float* mout = hout + (size_t)BSZ * NCC * NN * DD;     // (16,64,256,64)

  dim3 grid(BSZ * NCC);
  pack_weights<<<128, 256, 0, stream>>>(sw1, sw2, mw1, mw2, sw1b, sw2b, mw1b, mw2b);
  mod_kernel<<<grid, 64, 0, stream>>>(cc, dlog, bglog, modw1, modb1, modw2, modb2,
                                      decay, bgate);
  recv_gemm<<<grid, 512, 0, stream>>>(W, msg, recv);
  add_inj_border<<<grid, 256, 0, stream>>>(x, msg, iw, ib, c2g, bgate, recv);
  fused_update<<<grid, 256, 0, stream>>>(recv, h, decay, sw1b, sb1, sw2b, sb2,
                                         mw1b, mb1, mw2b, mb2, nid, rout, hout, mout);
}
